// CausalAttention_20504173871822
// MI455X (gfx1250) — compile-verified, matrix operand order checked by reading, not run on hardware
//
#include <hip/hip_runtime.h>
#include <hip/hip_bf16.h>
#include <stdint.h>

// ---------------------------------------------------------------------------
// CDNA5 (gfx1250) fused cross-attention:
//   prep: W -> W^T bf16 ; q = x@Wq ; kv = ctx@Wkv (bf16 out) ;
//   masked softmax over 32 keys/query (wave32-native) ; attn = w*V ;
//   out = attn@Wout + b
// GEMMs: v_wmma_f32_16x16x32_bf16, double-buffered LDS, TDM for B tiles.
// ---------------------------------------------------------------------------

#ifndef USE_TDM
#define USE_TDM 1
#endif

typedef __attribute__((ext_vector_type(16))) __bf16 v16bf;
typedef __attribute__((ext_vector_type(8)))  float  v8f;
typedef unsigned int u32x4 __attribute__((ext_vector_type(4)));
typedef int          i32x4 __attribute__((ext_vector_type(4)));
typedef int          i32x8 __attribute__((ext_vector_type(8)));

// Native f32 -> bf16 (RNE) via hardware convert; avoids integer-ALU emulation.
__device__ __forceinline__ uint16_t f2bf(float f) {
  union { __bf16 h; uint16_t u; } c;
  c.h = (__bf16)f;
  return c.u;
}
__device__ __forceinline__ uint32_t pk2bf(float lo, float hi) {
  return (uint32_t)f2bf(lo) | ((uint32_t)f2bf(hi) << 16);
}
__device__ __forceinline__ float bf2f(uint16_t h) {
  return __uint_as_float(((uint32_t)h) << 16);
}

union FragBF { uint32_t u32[8]; uint16_t u16[16]; v16bf v; };

#define BM 128
#define BN 128
#define BK 32
#define LDP 40   // padded LDS row (ushorts): 80B stride == TDM pad 16dw+4dw

// W [K][N] f32 -> Wt [N][K] bf16 (tiled transpose, both sides coalesced)
__global__ __launch_bounds__(256)
void transpose_bf16_kernel(const float* __restrict__ W, uint16_t* __restrict__ Wt,
                           int K, int N)
{
  __shared__ uint16_t tile[32][33];
  const int tx = threadIdx.x & 31;
  const int ty = threadIdx.x >> 5;           // 0..7
  #pragma unroll
  for (int r = 0; r < 4; ++r) {
    int k = blockIdx.y * 32 + ty + r * 8;
    int n = blockIdx.x * 32 + tx;
    tile[ty + r * 8][tx] = f2bf(W[(size_t)k * N + n]);
  }
  __syncthreads();
  #pragma unroll
  for (int r = 0; r < 4; ++r) {
    int n = blockIdx.x * 32 + ty + r * 8;
    int k = blockIdx.y * 32 + tx;
    Wt[(size_t)n * K + k] = tile[tx][ty + r * 8];
  }
}

// C[M,N] = A[M,K](f32, converted) * Bt[N,K](bf16, pre-transposed), f32 accum.
template<bool BF16OUT, bool HAS_BIAS>
__global__ __launch_bounds__(256)
void gemm_bf16_kernel(const float* __restrict__ A, const uint16_t* __restrict__ Bt,
                      float* __restrict__ Cf, uint16_t* __restrict__ Cb,
                      const float* __restrict__ bias,
                      int M, int N, int K)
{
  __shared__ uint16_t lA[2][BM][LDP];   // [m][k] bf16, double buffered
  __shared__ uint16_t lB[2][BN][LDP];   // [n][k] bf16, double buffered

  const int tid  = threadIdx.x;
  const int wave = tid >> 5;
  const int lane = tid & 31;
  const int l16  = lane & 15;
  const int lh   = lane >> 4;
  const int wm   = wave >> 1;
  const int wn   = wave & 1;
  const int rowbase  = wm * 32;
  const int colbase  = wn * 64;
  const int blockRow = blockIdx.y * BM;
  const int blockCol = blockIdx.x * BN;

  v8f acc[2][4];
  #pragma unroll
  for (int i = 0; i < 2; ++i)
    #pragma unroll
    for (int j = 0; j < 4; ++j)
      acc[i][j] = (v8f){0.f,0.f,0.f,0.f,0.f,0.f,0.f,0.f};

  int kk[8];
  #pragma unroll
  for (int j = 0; j < 8; ++j)
    kk[j] = ((j < 4) ? 2*j : 16 + 2*(j-4)) + lh*8;

  const int kSteps = K / BK;
  const int ar0 = tid >> 3;          // 0..31
  const int ac4 = (tid & 7) * 4;     // 0..28
  float4 ar[4];                      // A staging (next tile)

  auto loadA = [&](int ks) {
    #pragma unroll
    for (int r = 0; r < 4; ++r)
      ar[r] = *(const float4*)(A + (size_t)(blockRow + ar0 + r*32) * K + ks*BK + ac4);
  };
  auto storeA = [&](int buf) {
    #pragma unroll
    for (int r = 0; r < 4; ++r)
      *(uint2*)&lA[buf][ar0 + r*32][ac4] =
          make_uint2(pk2bf(ar[r].x, ar[r].y), pk2bf(ar[r].z, ar[r].w));
  };
  auto loadB = [&](int ks, int buf) {
#if USE_TDM
    if (wave == 0) {
      // Tensor DMA: 2D tile 32(k) x 128(n) of bf16 from Bt[N][K] into LDS,
      // padded 4 DWORDs per 16-DWORD row -> 80B LDS row stride (LDP=40).
      const uint16_t* gp = Bt + (size_t)blockCol * K + (size_t)ks * BK;
      uint64_t ga  = (uint64_t)(uintptr_t)gp;
      uint32_t lds = (uint32_t)(uintptr_t)&lB[buf][0][0];
      u32x4 g0 = { 1u,                                   // count=1, user mode
                   lds,                                  // lds_addr
                   (uint32_t)ga,                         // global_addr lo
                   (uint32_t)((ga >> 32) & 0x01FFFFFFu) | (2u << 30) };  // type=2
      i32x8 g1 = { (int)((1u<<16) | (1u<<20) | (3u<<22) | (3u<<25)),
                   //    data=2B    pad_en     intv=16dw   amount=4dw
                   (int)(((uint32_t)K & 0xFFFFu) << 16),                 // dim0 lo
                   (int)(((uint32_t)K >> 16) | (((uint32_t)N & 0xFFFFu) << 16)),
                   (int)(((uint32_t)N >> 16) | (32u << 16)),             // tile_dim0=32
                   (int)128,                                             // tile_dim1=128
                   (int)K,                                               // dim0_stride lo
                   0, 0 };
      i32x4 z4 = {0, 0, 0, 0};
#if __clang_major__ >= 23
      i32x8 z8 = {0, 0, 0, 0, 0, 0, 0, 0};
      __builtin_amdgcn_tensor_load_to_lds(g0, g1, z4, z4, z8, 0);
#else
      __builtin_amdgcn_tensor_load_to_lds(g0, g1, z4, z4, 0);
#endif
    }
#else
    #pragma unroll
    for (int r = 0; r < 4; ++r) {
      const uint16_t* gp = Bt + (size_t)(blockCol + ar0 + r*32) * K + (size_t)ks * BK + ac4;
      *(uint2*)&lB[buf][ar0 + r*32][ac4] = *(const uint2*)gp;
    }
#endif
  };

  // ---- pipeline prologue: tile 0 into buffer 0 ----
  int p = 0;
  loadB(0, 0);
  loadA(0);
  storeA(0);
  __builtin_amdgcn_s_wait_tensorcnt(0);
  __syncthreads();

  for (int ks = 0; ks < kSteps; ++ks) {
    if (ks + 1 < kSteps) {
      loadB(ks + 1, p ^ 1);   // TDM fills other buffer during compute
      loadA(ks + 1);          // global f32 loads in flight during WMMAs
    }
    if (ks + 2 < kSteps)
      __builtin_prefetch(A + (size_t)(blockRow + ar0) * K + (ks + 2) * BK + ac4, 0, 3);

    FragBF aF[2], bF[4];
    #pragma unroll
    for (int mi = 0; mi < 2; ++mi) {
      int row = rowbase + mi*16 + l16;
      #pragma unroll
      for (int j = 0; j < 8; ++j)
        aF[mi].u32[j] = *(const uint32_t*)&lA[p][row][kk[j]];
    }
    #pragma unroll
    for (int ni = 0; ni < 4; ++ni) {
      int col = colbase + ni*16 + l16;
      #pragma unroll
      for (int j = 0; j < 8; ++j)
        bF[ni].u32[j] = *(const uint32_t*)&lB[p][col][kk[j]];
    }
    #pragma unroll
    for (int mi = 0; mi < 2; ++mi)
      #pragma unroll
      for (int ni = 0; ni < 4; ++ni)
        acc[mi][ni] = __builtin_amdgcn_wmma_f32_16x16x32_bf16(
            false, aF[mi].v, false, bF[ni].v, (short)0, acc[mi][ni], false, false);

    if (ks + 1 < kSteps) {
      storeA(p ^ 1);
      __builtin_amdgcn_s_wait_tensorcnt(0);   // NOP when count==0
      __syncthreads();
      p ^= 1;
    }
  }

  // ---- epilogue: lane l16 = N, vgpr r = M (+8 for lanes 16..31) ----
  #pragma unroll
  for (int mi = 0; mi < 2; ++mi) {
    #pragma unroll
    for (int ni = 0; ni < 4; ++ni) {
      int gcol = blockCol + colbase + ni*16 + l16;
      float bv = 0.0f;
      if constexpr (HAS_BIAS) bv = bias[gcol];
      #pragma unroll
      for (int r = 0; r < 8; ++r) {
        int grow = blockRow + rowbase + mi*16 + r + 8*lh;
        float val = acc[mi][ni][r];
        if constexpr (HAS_BIAS) val += bv;
        if constexpr (BF16OUT) Cb[(size_t)grow * N + gcol] = f2bf(val);
        else                   Cf[(size_t)grow * N + gcol] = val;
      }
    }
  }
}

// One wave per (b,h,i). Lane t handles key j = i + 128t; softmax over 32 lanes.
__global__ __launch_bounds__(256)
void score_softmax_kernel(const float* __restrict__ q,
                          const uint16_t* __restrict__ kv,   // [b,4096,2048] bf16 (K|V)
                          float* __restrict__ wts)           // [b,h,128,32] f32
{
  const int lane = threadIdx.x & 31;
  const int idx  = blockIdx.x * 8 + (threadIdx.x >> 5);
  const int b   = idx >> 11;
  const int rem = idx & 2047;
  const int h   = rem >> 7;
  const int i   = rem & 127;
  const int t   = lane;

  const uint16_t* kp = kv + ((size_t)(b*4096 + t*128 + i)) * 2048 + h*64;
  const float*    qp = q  + ((size_t)(b*128 + i)) * 1024 + h*64;

  float s = 0.f;
  #pragma unroll
  for (int c4 = 0; c4 < 8; ++c4) {
    uint4  kb = ((const uint4*)kp)[c4];
    float4 q0 = ((const float4*)qp)[c4*2];
    float4 q1 = ((const float4*)qp)[c4*2 + 1];
    s += q0.x * bf2f((uint16_t)(kb.x & 0xFFFF)) + q0.y * bf2f((uint16_t)(kb.x >> 16));
    s += q0.z * bf2f((uint16_t)(kb.y & 0xFFFF)) + q0.w * bf2f((uint16_t)(kb.y >> 16));
    s += q1.x * bf2f((uint16_t)(kb.z & 0xFFFF)) + q1.y * bf2f((uint16_t)(kb.z >> 16));
    s += q1.z * bf2f((uint16_t)(kb.w & 0xFFFF)) + q1.w * bf2f((uint16_t)(kb.w >> 16));
  }
  s *= 0.125f;   // dim_head^-0.5

  float m = s;
  #pragma unroll
  for (int off = 16; off > 0; off >>= 1) m = fmaxf(m, __shfl_xor(m, off, 32));
  float e = __expf(s - m);
  float l = e;
  #pragma unroll
  for (int off = 16; off > 0; off >>= 1) l += __shfl_xor(l, off, 32);

  wts[(size_t)idx * 32 + t] = e / l;
}

// attn[b,i,n] = sum_t w[b,h(n),i,t] * V[b, i+128t, n]
__global__ __launch_bounds__(256)
void av_kernel(const float* __restrict__ wts, const uint16_t* __restrict__ kv,
               float* __restrict__ attn)
{
  int g   = blockIdx.x * 256 + threadIdx.x;
  int n   = g & 1023;
  int row = g >> 10;
  int b   = row >> 7;
  int i   = row & 127;
  int h   = n >> 6;
  const float*    wp = wts + ((size_t)((b*16 + h)*128 + i)) * 32;
  const uint16_t* vp = kv  + ((size_t)(b*4096 + i)) * 2048 + 1024 + n;
  float acc = 0.f;
  #pragma unroll
  for (int t = 0; t < 32; ++t)
    acc += wp[t] * bf2f(vp[(size_t)t * 128 * 2048]);
  attn[g] = acc;
}

extern "C" void kernel_launch(void* const* d_in, const int* in_sizes, int n_in,
                              void* d_out, int out_size, void* d_ws, size_t ws_size,
                              hipStream_t stream)
{
  const float* x    = (const float*)d_in[0];  // [16,128,1024]
  const float* ctx  = (const float*)d_in[1];  // [16,4096,1024]
  const float* Wq   = (const float*)d_in[2];  // [1024,1024]
  const float* Wkv  = (const float*)d_in[3];  // [1024,2048]
  const float* Wout = (const float*)d_in[4];  // [1024,1024]
  const float* bout = (const float*)d_in[5];  // [1024]
  float* out = (float*)d_out;                 // [16,128,1024]

  char* ws = (char*)d_ws;
  const size_t MiB = 1024 * 1024;
  float*    q     = (float*)   (ws);               //   8 MiB
  uint16_t* kv    = (uint16_t*)(ws + 8   * MiB);   // 256 MiB bf16
  float*    wts   = (float*)   (ws + 264 * MiB);   //   4 MiB
  float*    attn  = (float*)   (ws + 268 * MiB);   //   8 MiB
  uint16_t* WqT   = (uint16_t*)(ws + 276 * MiB);   //   2 MiB [1024][1024]
  uint16_t* WkvT  = (uint16_t*)(ws + 278 * MiB);   //   4 MiB [2048][1024]
  uint16_t* WoutT = (uint16_t*)(ws + 282 * MiB);   //   2 MiB [1024][1024]

  // 0) weights -> transposed bf16 (L2-resident, feeds TDM B-tile loads)
  transpose_bf16_kernel<<<dim3(1024/32, 1024/32), 256, 0, stream>>>(Wq,   WqT,   1024, 1024);
  transpose_bf16_kernel<<<dim3(2048/32, 1024/32), 256, 0, stream>>>(Wkv,  WkvT,  1024, 2048);
  transpose_bf16_kernel<<<dim3(1024/32, 1024/32), 256, 0, stream>>>(Wout, WoutT, 1024, 1024);

  // 1) q = x @ Wq
  gemm_bf16_kernel<false, false><<<dim3(1024/BN, 2048/BM), 256, 0, stream>>>(
      x, WqT, q, nullptr, nullptr, 2048, 1024, 1024);

  // 2) kv = context @ Wkv (dominant GEMM, bf16 out)
  gemm_bf16_kernel<true, false><<<dim3(2048/BN, 65536/BM), 256, 0, stream>>>(
      ctx, WkvT, nullptr, kv, nullptr, 65536, 2048, 1024);

  // 3) masked scores + softmax (32 keys/query)
  score_softmax_kernel<<<dim3(32768/8), 256, 0, stream>>>(q, kv, wts);

  // 4) weighted sum of 32 V rows
  av_kernel<<<dim3((2048*1024)/256), 256, 0, stream>>>(wts, kv, attn);

  // 5) out = attn @ Wout + bout
  gemm_bf16_kernel<false, true><<<dim3(1024/BN, 2048/BM), 256, 0, stream>>>(
      attn, WoutT, out, nullptr, bout, 2048, 1024, 1024);

  (void)in_sizes; (void)n_in; (void)out_size; (void)ws_size;
}
